// FusedMoE_44848048505292
// MI455X (gfx1250) — matrix operand order repaired and checked by reading, not compiled, not run on hardware
//
#include <hip/hip_runtime.h>
#include <hip/hip_bf16.h>

// ---------------------------------------------------------------------------
// Fused MoE (top-2 of 8 experts, SwiGLU) for MI455X / gfx1250.
// Compute-bound (AI ~800 FLOP/B >> roofline knee ~100 FLOP/B) -> bf16 WMMA:
//   v_wmma_f32_16x16x32_bf16, fp32 accumulate.
// Weights are pre-converted fp32->bf16 once per launch (~12 us of HBM traffic
// vs ~80 us of compute), so ALL GEMM operands are bf16 and the 128x32 B tiles
// are staged with the Tensor Data Mover (tensor_load_to_lds) using D# LDS
// padding to produce the bank-spread KCP=40 stride. Double-buffered LDS:
// wave 0 issues next chunk's TDM while all waves run WMMA on current chunk,
// then s_wait_tensorcnt + barrier publishes it. SwiGLU uses v_rcp_f32 (output
// is bf16; 1-ulp rcp is exact enough) instead of the IEEE divide sequence.
// ---------------------------------------------------------------------------

#define T_TOK 8192
#define HID   1024
#define FFN_  2048
#define NEXP  8
#define TOPK  2
#define TKA   (T_TOK * TOPK)   // total assignment rows = 16384

#define KC  32                 // K-chunk per LDS stage (one WMMA k-step)
#define KCP 40                 // padded LDS row stride (halves): 80 B rows

typedef __attribute__((ext_vector_type(16))) __bf16 v16bf;
typedef __attribute__((ext_vector_type(8)))  __bf16 v8bf;
typedef __attribute__((ext_vector_type(4)))  __bf16 v4bf;
typedef __attribute__((ext_vector_type(8)))  float  v8f;
typedef __attribute__((ext_vector_type(4)))  unsigned int u32x4;
typedef __attribute__((ext_vector_type(8)))  int i32x8;
typedef __attribute__((ext_vector_type(4)))  int i32x4;

// ---- workspace layout (bytes) ---------------------------------------------
#define WS_COUNTS 0
#define WS_OFFS   64
#define WS_CURS   128
#define WS_TOKID  256
#define WS_TOKW   (WS_TOKID + TKA * 4)
#define WS_ACT    ((size_t)1 << 20)                       // act bf16: 64 MB
#define WS_XB     (WS_ACT + (size_t)TKA * FFN_ * 2)       // x bf16:  16 MB
#define WS_GUPB   (WS_XB + (size_t)T_TOK * HID * 2)       // Wgu bf16: 64 MB
#define WS_DWNB   (WS_GUPB + (size_t)NEXP * 2 * FFN_ * HID * 2) // Wd: 32 MB

#if defined(__gfx1250__) && __has_builtin(__builtin_amdgcn_tensor_load_to_lds)
#define USE_TDM 1
#else
#define USE_TDM 0
#endif
#if defined(__gfx1250__) && __has_builtin(__builtin_amdgcn_s_wait_tensorcnt)
#define TWAIT() __builtin_amdgcn_s_wait_tensorcnt(0)
#else
#define TWAIT()
#endif

__device__ __forceinline__ float fast_rcp(float x) {
#if defined(__gfx1250__) && __has_builtin(__builtin_amdgcn_rcpf)
  return __builtin_amdgcn_rcpf(x);     // v_rcp_f32: ample for bf16 output
#else
  return 1.0f / x;
#endif
}

// ---- fragment loaders (ISA 7.12.2 16-bit layouts) -------------------------
// A (16x32): lanes 0-15 -> M=lane, K={0..7,16..23}; lanes 16-31 -> M=lane-16,
// K={8..15,24..31}.
__device__ __forceinline__ v16bf load_a_frag(const __bf16* tile, int lane) {
  const int r  = lane & 15;
  const int kk = (lane >> 4) << 3;                   // 0 or 8
  const __bf16* base = tile + r * KCP + kk;
  v8bf lo = *(const v8bf*)(base);
  v8bf hi = *(const v8bf*)(base + 16);
  v16bf a;
#pragma unroll
  for (int i = 0; i < 8; ++i) { a[i] = lo[i]; a[i + 8] = hi[i]; }
  return a;
}
// B (32x16) from row-major W[N][K]: lanes 0-15 -> N=lane, K={0..7,16..23};
// lanes 16-31 -> N=lane-16, K={8..15,24..31}.
__device__ __forceinline__ v16bf load_b_frag(const __bf16* tile, int lane) {
  const int n  = lane & 15;
  const int kk = (lane >> 4) << 3;                   // 0 or 8
  const __bf16* base = tile + n * KCP + kk;
  v8bf lo = *(const v8bf*)(base);
  v8bf hi = *(const v8bf*)(base + 16);
  v16bf b;
#pragma unroll
  for (int i = 0; i < 8; ++i) { b[i] = lo[i]; b[i + 8] = hi[i]; }
  return b;
}

#define WMMA_BF16(A, B, C) \
  __builtin_amdgcn_wmma_f32_16x16x32_bf16(false, (A), false, (B), (short)0, (C), false, false)

#if USE_TDM
// Build D# group1 for a dense 2D bf16 tile: 128 rows x KC cols out of a
// row-major tensor with row length Ktot. LDS padding: after every 16 DWORDs
// (=64 B = one KC row) insert 4 DWORDs (16 B) -> 80 B rows == KCP stride.
__device__ __forceinline__ i32x8 tdm_g1(int Ktot) {
  i32x8 g1;
  g1[0] = (int)((1u << 16)      // data_size = 2 bytes
              | (1u << 20)      // pad_enable
              | (3u << 22)      // pad_interval: 16 DWORDs
              | (3u << 25));    // pad_amount: 4 DWORDs
  g1[1] = (int)(((unsigned)Ktot & 0xFFFFu) << 16);                  // dim0 lo
  g1[2] = (int)((((unsigned)Ktot >> 16) & 0xFFFFu) | (128u << 16)); // dim0 hi | dim1 lo
  g1[3] = (int)((unsigned)KC << 16);                                // dim1 hi=0 | tile_dim0
  g1[4] = 128;                                                      // tile_dim1
  g1[5] = Ktot;                                                     // dim0_stride lo
  g1[6] = 0;
  g1[7] = 0;
  return g1;
}
__device__ __forceinline__ void tdm_issue(unsigned lds_addr,
                                          unsigned long long gaddr,
                                          i32x8 g1) {
  u32x4 g0;
  g0[0] = 1u;                                        // count=1
  g0[1] = lds_addr;
  g0[2] = (unsigned)gaddr;                           // global_addr[31:0]
  g0[3] = (unsigned)((gaddr >> 32) & 0x1FFFFFFu)     // global_addr[56:32]
        | (2u << 30);                                // type = 2 (image)
  i32x4 z4 = {0, 0, 0, 0};
#if __clang_major__ >= 23
  i32x8 z8 = {0, 0, 0, 0, 0, 0, 0, 0};
  __builtin_amdgcn_tensor_load_to_lds(g0, g1, z4, z4, z8, 0);
#else
  __builtin_amdgcn_tensor_load_to_lds(g0, g1, z4, z4, 0);
#endif
}
#endif

// ---- phase 0: zero output, convert x -> bf16, zero counters ---------------
__global__ __launch_bounds__(256) void moe_init(const float* __restrict__ hs,
                                                float* __restrict__ out,
                                                __bf16* __restrict__ xb,
                                                int* __restrict__ counts) {
  const int i = blockIdx.x * 256 + threadIdx.x;
  if (i < T_TOK * HID) {
    out[i] = 0.0f;
    xb[i]  = (__bf16)hs[i];
  }
  if (i < NEXP) counts[i] = 0;
}

// fp32 -> bf16 weight pre-conversion (4 elements/thread)
__global__ __launch_bounds__(256) void moe_cvt4(const float4* __restrict__ s,
                                                v4bf* __restrict__ d, int n4) {
  const int i = blockIdx.x * 256 + threadIdx.x;
  if (i < n4) {
    const float4 f = s[i];
    v4bf h;
    h[0] = (__bf16)f.x; h[1] = (__bf16)f.y; h[2] = (__bf16)f.z; h[3] = (__bf16)f.w;
    d[i] = h;
  }
}

__global__ __launch_bounds__(256) void moe_count(const int* __restrict__ ids,
                                                 int* __restrict__ counts) {
  const int i = blockIdx.x * 256 + threadIdx.x;
  if (i < TKA) atomicAdd(&counts[ids[i]], 1);
}

__global__ void moe_scan(const int* __restrict__ counts, int* __restrict__ offs,
                         int* __restrict__ curs) {
  if (threadIdx.x == 0 && blockIdx.x == 0) {
    int s = 0;
    for (int e = 0; e < NEXP; ++e) { offs[e] = s; curs[e] = s; s += counts[e]; }
    offs[NEXP] = s;
  }
}

__global__ __launch_bounds__(256) void moe_fill(const int* __restrict__ ids,
                                                const float* __restrict__ wts,
                                                int* __restrict__ curs,
                                                int* __restrict__ tokid,
                                                float* __restrict__ tokw) {
  const int i = blockIdx.x * 256 + threadIdx.x;
  if (i < TKA) {
    const int e = ids[i];
    const int p = atomicAdd(&curs[e], 1);
    tokid[p] = i / TOPK;
    tokw[p]  = wts[i];
  }
}

// ---- phase 1: gu = x . Wgu^T, fused SwiGLU, act -> bf16 -------------------
// grid (F/128, 256 worst-case M-tiles, E); 8 waves arranged 2(M) x 4(N):
// each wave owns 32 rows x 32 cols of gate AND up (8 v8f accumulators).
__global__ __launch_bounds__(256)
void moe_gemm1(const __bf16* __restrict__ gupb, const __bf16* __restrict__ xb,
               __bf16* __restrict__ act, const int* __restrict__ offs,
               const int* __restrict__ tokid) {
  const int e   = blockIdx.z;
  const int seg = offs[e];
  const int cnt = offs[e + 1] - seg;
  const int m0  = blockIdx.y * 64;
  if (m0 >= cnt) return;
  const int n0  = blockIdx.x * 128;

  __shared__ __bf16 Ab[2][64 * KCP];
  __shared__ __bf16 Bg[2][128 * KCP];
  __shared__ __bf16 Bu[2][128 * KCP];
  __shared__ int    stok[64];

  const int tid  = threadIdx.x;
  const int lane = tid & 31;
  const int wv   = tid >> 5;
  const int Mw   = wv >> 2;        // 0..1 -> rows 32*Mw
  const int Nw   = wv & 3;         // 0..3 -> cols 32*Nw

  if (tid < 64) {
    int r = m0 + tid;
    if (r >= cnt) r = cnt - 1;     // clamp: loads valid, stores masked
    stok[tid] = tokid[seg + r];
  }
  __syncthreads();

  v8f accG[2][2] = {}, accU[2][2] = {};

  // A staging role: 64x32 bf16, one 16B chunk per thread.
  const int arow = tid >> 2, acol = (tid & 3) * 8;
  const __bf16* asrc = xb + (size_t)stok[arow] * HID + acol;

  const __bf16* bg0 = gupb + ((size_t)e * 2 * FFN_ + n0) * HID;
  const __bf16* bu0 = bg0 + (size_t)FFN_ * HID;
#if USE_TDM
  const i32x8 g1 = tdm_g1(HID);
#else
  const __bf16* gsrc = bg0 + (size_t)(tid >> 2) * HID + acol;
  const __bf16* usrc = gsrc + (size_t)FFN_ * HID;
#endif

  // ---- prologue: stage chunk 0 into buffer 0 ----
  *(v8bf*)&Ab[0][arow * KCP + acol] = *(const v8bf*)(asrc);
#if USE_TDM
  if (wv == 0) {
    tdm_issue((unsigned)(uintptr_t)&Bg[0][0], (unsigned long long)(uintptr_t)bg0, g1);
    tdm_issue((unsigned)(uintptr_t)&Bu[0][0], (unsigned long long)(uintptr_t)bu0, g1);
    TWAIT();
  }
#else
  *(v8bf*)&Bg[0][(tid >> 2) * KCP + acol]        = *(const v8bf*)(gsrc);
  *(v8bf*)&Bg[0][((tid >> 2) + 64) * KCP + acol] = *(const v8bf*)(gsrc + (size_t)64 * HID);
  *(v8bf*)&Bu[0][(tid >> 2) * KCP + acol]        = *(const v8bf*)(usrc);
  *(v8bf*)&Bu[0][((tid >> 2) + 64) * KCP + acol] = *(const v8bf*)(usrc + (size_t)64 * HID);
#endif
  __syncthreads();

  int b = 0;
  for (int k0 = 0; k0 < HID; k0 += KC) {
    const int kn = k0 + KC;
    if (kn < HID) {                // stage next chunk into other buffer
      *(v8bf*)&Ab[b ^ 1][arow * KCP + acol] = *(const v8bf*)(asrc + kn);
#if USE_TDM
      if (wv == 0) {
        tdm_issue((unsigned)(uintptr_t)&Bg[b ^ 1][0],
                  (unsigned long long)(uintptr_t)bg0 + (unsigned long long)kn * 2, g1);
        tdm_issue((unsigned)(uintptr_t)&Bu[b ^ 1][0],
                  (unsigned long long)(uintptr_t)bu0 + (unsigned long long)kn * 2, g1);
      }
#else
      *(v8bf*)&Bg[b ^ 1][(tid >> 2) * KCP + acol]        = *(const v8bf*)(gsrc + kn);
      *(v8bf*)&Bg[b ^ 1][((tid >> 2) + 64) * KCP + acol] = *(const v8bf*)(gsrc + (size_t)64 * HID + kn);
      *(v8bf*)&Bu[b ^ 1][(tid >> 2) * KCP + acol]        = *(const v8bf*)(usrc + kn);
      *(v8bf*)&Bu[b ^ 1][((tid >> 2) + 64) * KCP + acol] = *(const v8bf*)(usrc + (size_t)64 * HID + kn);
#endif
    }

    const __bf16* Abase = &Ab[b][(32 * Mw) * KCP];
    const v16bf a0 = load_a_frag(Abase, lane);
    const v16bf a1 = load_a_frag(Abase + 16 * KCP, lane);
#pragma unroll
    for (int j = 0; j < 2; ++j) {
      const v16bf bg = load_b_frag(&Bg[b][(32 * Nw + 16 * j) * KCP], lane);
      accG[0][j] = WMMA_BF16(a0, bg, accG[0][j]);
      accG[1][j] = WMMA_BF16(a1, bg, accG[1][j]);
      const v16bf bu = load_b_frag(&Bu[b][(32 * Nw + 16 * j) * KCP], lane);
      accU[0][j] = WMMA_BF16(a0, bu, accU[0][j]);
      accU[1][j] = WMMA_BF16(a1, bu, accU[1][j]);
    }
#if USE_TDM
    if (wv == 0) TWAIT();
#endif
    __syncthreads();
    b ^= 1;
  }

  // epilogue: SwiGLU (rcp-based sigmoid), store bf16 activations.
  const int mloc = (lane < 16) ? 0 : 8;
  const int nl   = lane & 15;
#pragma unroll
  for (int mi = 0; mi < 2; ++mi) {
#pragma unroll
    for (int j = 0; j < 2; ++j) {
#pragma unroll
      for (int i = 0; i < 8; ++i) {
        const int row = m0 + 32 * Mw + 16 * mi + i + mloc;
        if (row < cnt) {
          const float g = accG[mi][j][i];
          const float u = accU[mi][j][i];
          const float s = g * fast_rcp(1.0f + __expf(-g));
          act[(size_t)(seg + row) * FFN_ + (n0 + 32 * Nw + 16 * j + nl)] = (__bf16)(s * u);
        }
      }
    }
  }
}

// ---- phase 2: d = act . Wdown^T, scale, atomic combine --------------------
__global__ __launch_bounds__(256)
void moe_gemm2(const __bf16* __restrict__ dwb, const __bf16* __restrict__ act,
               float* __restrict__ out, const int* __restrict__ offs,
               const int* __restrict__ tokid, const float* __restrict__ tokw) {
  const int e   = blockIdx.z;
  const int seg = offs[e];
  const int cnt = offs[e + 1] - seg;
  const int m0  = blockIdx.y * 64;
  if (m0 >= cnt) return;
  const int n0  = blockIdx.x * 128;

  __shared__ __bf16 Ab[2][64 * KCP];
  __shared__ __bf16 Bd[2][128 * KCP];
  __shared__ int    stok[64];
  __shared__ float  sw[64];

  const int tid  = threadIdx.x;
  const int lane = tid & 31;
  const int wv   = tid >> 5;
  const int Mw   = wv >> 2;
  const int Nw   = wv & 3;

  if (tid < 64) {
    int r = m0 + tid;
    if (r >= cnt) r = cnt - 1;
    stok[tid] = tokid[seg + r];
    sw[tid]   = tokw[seg + r];
  }
  __syncthreads();

  v8f acc[2][2] = {};

  const int arow = tid >> 2, acol = (tid & 3) * 8;
  int ar = m0 + arow; if (ar >= cnt) ar = cnt - 1;
  const __bf16* asrc = act + (size_t)(seg + ar) * FFN_ + acol;

  const __bf16* bd0 = dwb + ((size_t)e * HID + n0) * FFN_;
#if USE_TDM
  const i32x8 g1 = tdm_g1(FFN_);
#else
  const __bf16* dsrc = bd0 + (size_t)(tid >> 2) * FFN_ + acol;
#endif

  *(v8bf*)&Ab[0][arow * KCP + acol] = *(const v8bf*)(asrc);
#if USE_TDM
  if (wv == 0) {
    tdm_issue((unsigned)(uintptr_t)&Bd[0][0], (unsigned long long)(uintptr_t)bd0, g1);
    TWAIT();
  }
#else
  *(v8bf*)&Bd[0][(tid >> 2) * KCP + acol]        = *(const v8bf*)(dsrc);
  *(v8bf*)&Bd[0][((tid >> 2) + 64) * KCP + acol] = *(const v8bf*)(dsrc + (size_t)64 * FFN_);
#endif
  __syncthreads();

  int b = 0;
  for (int k0 = 0; k0 < FFN_; k0 += KC) {
    const int kn = k0 + KC;
    if (kn < FFN_) {
      *(v8bf*)&Ab[b ^ 1][arow * KCP + acol] = *(const v8bf*)(asrc + kn);
#if USE_TDM
      if (wv == 0)
        tdm_issue((unsigned)(uintptr_t)&Bd[b ^ 1][0],
                  (unsigned long long)(uintptr_t)bd0 + (unsigned long long)kn * 2, g1);
#else
      *(v8bf*)&Bd[b ^ 1][(tid >> 2) * KCP + acol]        = *(const v8bf*)(dsrc + kn);
      *(v8bf*)&Bd[b ^ 1][((tid >> 2) + 64) * KCP + acol] = *(const v8bf*)(dsrc + (size_t)64 * FFN_ + kn);
#endif
    }

    const __bf16* Abase = &Ab[b][(32 * Mw) * KCP];
    const v16bf a0 = load_a_frag(Abase, lane);
    const v16bf a1 = load_a_frag(Abase + 16 * KCP, lane);
#pragma unroll
    for (int j = 0; j < 2; ++j) {
      const v16bf bd = load_b_frag(&Bd[b][(32 * Nw + 16 * j) * KCP], lane);
      acc[0][j] = WMMA_BF16(a0, bd, acc[0][j]);
      acc[1][j] = WMMA_BF16(a1, bd, acc[1][j]);
    }
#if USE_TDM
    if (wv == 0) TWAIT();
#endif
    __syncthreads();
    b ^= 1;
  }

  const int mloc = (lane < 16) ? 0 : 8;
  const int nl   = lane & 15;
#pragma unroll
  for (int mi = 0; mi < 2; ++mi) {
#pragma unroll
    for (int j = 0; j < 2; ++j) {
#pragma unroll
      for (int i = 0; i < 8; ++i) {
        const int lrow = 32 * Mw + 16 * mi + i + mloc;
        const int row  = m0 + lrow;
        if (row < cnt) {
          const float v = acc[mi][j][i] * sw[lrow];
          atomicAdd(&out[(size_t)stok[lrow] * HID + (n0 + 32 * Nw + 16 * j + nl)], v);
        }
      }
    }
  }
}

// ---------------------------------------------------------------------------
extern "C" void kernel_launch(void* const* d_in, const int* in_sizes, int n_in,
                              void* d_out, int out_size, void* d_ws, size_t ws_size,
                              hipStream_t stream) {
  (void)in_sizes; (void)n_in; (void)out_size; (void)ws_size;
  const float* hs   = (const float*)d_in[0];   // [T, H]
  const float* tw   = (const float*)d_in[1];   // [T, K]
  const int*   tids = (const int*)d_in[2];     // [T, K]
  const float* gup  = (const float*)d_in[3];   // [E, 2F, H]
  const float* dwn  = (const float*)d_in[4];   // [E, H, F]
  float*       out  = (float*)d_out;           // [T, H]

  char*   ws     = (char*)d_ws;
  int*    counts = (int*)(ws + WS_COUNTS);
  int*    offs   = (int*)(ws + WS_OFFS);
  int*    curs   = (int*)(ws + WS_CURS);
  int*    tokid  = (int*)(ws + WS_TOKID);
  float*  tokw   = (float*)(ws + WS_TOKW);
  __bf16* actb   = (__bf16*)(ws + WS_ACT);
  __bf16* xb     = (__bf16*)(ws + WS_XB);
  __bf16* gupb   = (__bf16*)(ws + WS_GUPB);
  __bf16* dwnb   = (__bf16*)(ws + WS_DWNB);

  moe_init<<<(T_TOK * HID + 255) / 256, 256, 0, stream>>>(hs, out, xb, counts);

  const int ng4 = NEXP * 2 * FFN_ * HID / 4;   // 8,388,608
  const int nd4 = NEXP * HID * FFN_ / 4;       // 4,194,304
  moe_cvt4<<<(ng4 + 255) / 256, 256, 0, stream>>>((const float4*)gup, (v4bf*)gupb, ng4);
  moe_cvt4<<<(nd4 + 255) / 256, 256, 0, stream>>>((const float4*)dwn, (v4bf*)dwnb, nd4);

  moe_count<<<(TKA + 255) / 256, 256, 0, stream>>>(tids, counts);
  moe_scan<<<1, 32, 0, stream>>>(counts, offs, curs);
  moe_fill<<<(TKA + 255) / 256, 256, 0, stream>>>(tids, tw, curs, tokid, tokw);

  dim3 g1(FFN_ / 128, TKA / 64, NEXP);   // (16, 256, 8), dead tiles early-exit
  moe_gemm1<<<g1, 256, 0, stream>>>(gupb, xb, actb, offs, tokid);

  dim3 g2(HID / 128, TKA / 64, NEXP);    // (8, 256, 8)
  moe_gemm2<<<g2, 256, 0, stream>>>(dwnb, actb, out, offs, tokid, tokw);
}
